// BahdanauDecoder_29059748725660
// MI455X (gfx1250) — matrix operand order repaired and checked by reading, not compiled, not run on hardware
//
#include <hip/hip_runtime.h>
#include <hip/hip_bf16.h>
#include <math.h>

#define Hdim 1024
#define Tdim 4096
#define Bdim 10
#define Vdim 50257
#define Mdim (Tdim * Bdim)   // 40960 rows of (T,B,H) flattened
#define H3   (3 * Hdim)

typedef __attribute__((ext_vector_type(16))) __bf16 bf16x16;
typedef __attribute__((ext_vector_type(8)))  float  f32x8;
typedef __attribute__((ext_vector_type(4)))  float  f32x4;

// ---------------------------------------------------------------------------
// CDNA5 async global->LDS copies (ASYNCcnt-tracked).  LDS address = low 32
// bits of the generic pointer.
// ---------------------------------------------------------------------------
#define ASYNC_LDS_B64(lptr, gptr)                                         \
  asm volatile("global_load_async_to_lds_b64 %0, %1, off" ::              \
               "v"((unsigned)(unsigned long long)(lptr)),                 \
               "v"((unsigned long long)(gptr)) : "memory")

#define ASYNC_LDS_B128(lptr, gptr)                                        \
  asm volatile("global_load_async_to_lds_b128 %0, %1, off" ::             \
               "v"((unsigned)(unsigned long long)(lptr)),                 \
               "v"((unsigned long long)(gptr)) : "memory")

#define WAIT_ASYNCCNT_0() asm volatile("s_wait_asynccnt 0x0" ::: "memory")
#define WAIT_ASYNCCNT_1() asm volatile("s_wait_asynccnt 0x1" ::: "memory")

// ---------------------------------------------------------------------------
// A-fragment for V_WMMA_F32_16X16X32_BF16 (16-bit A 16x32, this code's contiguous K order, matched by B):
// lanes 0-15: row M=lane, K = kb..kb+15 with kb=0;
// lanes 16-31: row M=lane-16, kb=16.  Arow points at 32 fp32 K-values.
// ---------------------------------------------------------------------------
__device__ __forceinline__ bf16x16 make_a_frag(const float* Arow, int half) {
  const int kb = half * 16;
  f32x4 a0 = *(const f32x4*)(Arow + kb);
  f32x4 a1 = *(const f32x4*)(Arow + kb + 4);
  f32x4 a2 = *(const f32x4*)(Arow + kb + 8);
  f32x4 a3 = *(const f32x4*)(Arow + kb + 12);
  bf16x16 af;
  af[0]  = (__bf16)a0[0]; af[1]  = (__bf16)a0[1]; af[2]  = (__bf16)a0[2]; af[3]  = (__bf16)a0[3];
  af[4]  = (__bf16)a1[0]; af[5]  = (__bf16)a1[1]; af[6]  = (__bf16)a1[2]; af[7]  = (__bf16)a1[3];
  af[8]  = (__bf16)a2[0]; af[9]  = (__bf16)a2[1]; af[10] = (__bf16)a2[2]; af[11] = (__bf16)a2[3];
  af[12] = (__bf16)a3[0]; af[13] = (__bf16)a3[1]; af[14] = (__bf16)a3[2]; af[15] = (__bf16)a3[3];
  return af;
}

// 16 consecutive fp32 -> bf16x16 (B fragment: lane = column N, 16 consecutive K)
__device__ __forceinline__ bf16x16 cvt16_consecutive(const float* __restrict__ p) {
  f32x4 a0 = *(const f32x4*)(p);
  f32x4 a1 = *(const f32x4*)(p + 4);
  f32x4 a2 = *(const f32x4*)(p + 8);
  f32x4 a3 = *(const f32x4*)(p + 12);
  bf16x16 bf;
  bf[0]  = (__bf16)a0[0]; bf[1]  = (__bf16)a0[1]; bf[2]  = (__bf16)a0[2]; bf[3]  = (__bf16)a0[3];
  bf[4]  = (__bf16)a1[0]; bf[5]  = (__bf16)a1[1]; bf[6]  = (__bf16)a1[2]; bf[7]  = (__bf16)a1[3];
  bf[8]  = (__bf16)a2[0]; bf[9]  = (__bf16)a2[1]; bf[10] = (__bf16)a2[2]; bf[11] = (__bf16)a2[3];
  bf[12] = (__bf16)a3[0]; bf[13] = (__bf16)a3[1]; bf[14] = (__bf16)a3[2]; bf[15] = (__bf16)a3[3];
  return bf;
}

// fc_e_w fp32 -> bf16 (row-major HxH), read once per launch
__global__ __launch_bounds__(256) void k_convert_w(const float* __restrict__ src,
                                                   unsigned short* __restrict__ dst) {
  int i = blockIdx.x * 256 + threadIdx.x;
  ((__bf16*)dst)[i] = (__bf16)src[i];
}

// h = decoder_hidden[0] @ fc_h_w^T + fc_h_b   (B x H), wave-per-output dot
__global__ __launch_bounds__(256) void k_fc_hidden(const float* __restrict__ dec_h,
                                                   const float* __restrict__ w,
                                                   const float* __restrict__ bias,
                                                   float* __restrict__ out) {
  const int lane = threadIdx.x & 31;
  const int wid  = blockIdx.x * 8 + (threadIdx.x >> 5);
  if (wid >= Bdim * Hdim) return;
  const int b = wid / Hdim, j = wid % Hdim;
  const float* x  = dec_h + b * Hdim;
  const float* wr = w + (size_t)j * Hdim;
  float acc = 0.f;
  for (int k = lane; k < Hdim; k += 32) acc += x[k] * wr[k];
  acc += __shfl_xor(acc, 1);  acc += __shfl_xor(acc, 2);
  acc += __shfl_xor(acc, 4);  acc += __shfl_xor(acc, 8);
  acc += __shfl_xor(acc, 16);
  if (lane == 0) out[wid] = acc + bias[j];
}

// ---------------------------------------------------------------------------
// Main GEMM: enc = encoder_output @ fc_e_w^T + fc_e_b  (M=40960, N=K=1024).
// A tile (16x32 fp32, shared by all 8 waves) staged into LDS with async
// global->LDS copies, double-buffered across k-steps so the fill of step k+1
// overlaps the WMMAs of step k.  B (bf16 weights) streamed from global.
// Fused epilogue: store enc + tanh-score partials.
// ---------------------------------------------------------------------------
__global__ __launch_bounds__(256)
void k_gemm_enc(const float* __restrict__ A,            // (Mdim, H)
                const unsigned short* __restrict__ Wb,  // (H, H) bf16
                const float* __restrict__ bias,         // fc_e_b
                const float* __restrict__ hvec,         // (B, H)
                const float* __restrict__ attn_w,       // (H)
                float* __restrict__ enc,                // (Mdim, H)
                float* __restrict__ scores_part) {      // (Mdim, 16)
  __shared__ __align__(16) float Abuf[2][16][32];   // 2 x 2KB double buffer

  const int tid  = threadIdx.x;
  const int lane = tid & 31;
  const int wave = tid >> 5;
  const int half = lane >> 4;
  const int l16  = lane & 15;
  const int m0 = blockIdx.y * 16;
  const int n0 = blockIdx.x * 512 + wave * 64;

  const __bf16* Wbb = (const __bf16*)Wb;

  // async stage of A[m0..m0+15][k0..k0+31]: 256 threads x 8B = 2KB
  const int srow = tid >> 4;          // 0..15
  const int spos = (tid & 15) * 2;    // 0,2,..,30
  const float* gArow = A + (size_t)(m0 + srow) * Hdim + spos;
  {
    const float* lp = &Abuf[0][srow][spos];
    ASYNC_LDS_B64(lp, gArow + 0);
  }

  f32x8 c[4] = {f32x8{0}, f32x8{0}, f32x8{0}, f32x8{0}};

  const int NI = Hdim / 32;  // 32 k-steps
  for (int i = 0; i < NI; ++i) {
    const int k0  = i * 32;
    const int cur = i & 1;
    if (i + 1 < NI) {
      const float* lp = &Abuf[cur ^ 1][srow][spos];
      ASYNC_LDS_B64(lp, gArow + (k0 + 32));
      WAIT_ASYNCCNT_1();   // in-order: fill of `cur` complete
    } else {
      WAIT_ASYNCCNT_0();
    }
    __syncthreads();       // all waves' fills of `cur` visible

    // issue streaming B loads first, then build A frag from LDS
    const int koff = k0 + half * 16;  // lane n = K 0..15, lane n+16 = K 16..31
    bf16x16 bf0 = *(const bf16x16*)(Wbb + (size_t)(n0 +  0 + l16) * Hdim + koff);
    bf16x16 bf1 = *(const bf16x16*)(Wbb + (size_t)(n0 + 16 + l16) * Hdim + koff);
    bf16x16 bf2 = *(const bf16x16*)(Wbb + (size_t)(n0 + 32 + l16) * Hdim + koff);
    bf16x16 bf3 = *(const bf16x16*)(Wbb + (size_t)(n0 + 48 + l16) * Hdim + koff);
    bf16x16 af = make_a_frag(&Abuf[cur][l16][0], half);

    c[0] = __builtin_amdgcn_wmma_f32_16x16x32_bf16(false, af, false, bf0, (short)0, c[0], false, false);
    c[1] = __builtin_amdgcn_wmma_f32_16x16x32_bf16(false, af, false, bf1, (short)0, c[1], false, false);
    c[2] = __builtin_amdgcn_wmma_f32_16x16x32_bf16(false, af, false, bf2, (short)0, c[2], false, false);
    c[3] = __builtin_amdgcn_wmma_f32_16x16x32_bf16(false, af, false, bf3, (short)0, c[3], false, false);

    __syncthreads();       // reads of `cur` done before it is refilled
  }

  // Epilogue: C/D layout — VGPR r: lanes 0-15 = row m0+r, lanes 16-31 = m0+8+r
  const int row_off = half * 8;
  float rowsum[8];
#pragma unroll
  for (int r = 0; r < 8; ++r) rowsum[r] = 0.f;

#pragma unroll
  for (int t = 0; t < 4; ++t) {
    const int col  = n0 + t * 16 + l16;
    const float bc = bias[col];
    const float aw = attn_w[col];
#pragma unroll
    for (int r = 0; r < 8; ++r) {
      const int row   = m0 + row_off + r;
      const float val = c[t][r] + bc;
      enc[(size_t)row * Hdim + col] = val;
      const int b = row % Bdim;   // row = t*B + b
      rowsum[r] += tanhf(hvec[b * Hdim + col] + val) * aw;
    }
  }

  const int chunk = blockIdx.x * 8 + wave;  // 0..15
#pragma unroll
  for (int r = 0; r < 8; ++r) {
    float s = rowsum[r];
    s += __shfl_xor(s, 1); s += __shfl_xor(s, 2);
    s += __shfl_xor(s, 4); s += __shfl_xor(s, 8);
    if (l16 == 0) {  // lane 0 -> row m0+r, lane 16 -> row m0+8+r
      const int row = m0 + row_off + r;
      scores_part[(size_t)row * 16 + chunk] = s;
    }
  }
}

// scores[b][t] = sum over 16 chunks of scores_part[t*B+b][c]
__global__ __launch_bounds__(256) void k_scores_reduce(const float* __restrict__ sp,
                                                       float* __restrict__ scores) {
  int m = blockIdx.x * 256 + threadIdx.x;
  if (m >= Mdim) return;
  float s = 0.f;
#pragma unroll
  for (int c = 0; c < 16; ++c) s += sp[(size_t)m * 16 + c];
  const int t = m / Bdim, b = m % Bdim;
  scores[(size_t)b * Tdim + t] = s;
}

// softmax over T, one block per batch element
__global__ __launch_bounds__(256) void k_softmax_T(const float* __restrict__ scores,
                                                   float* __restrict__ attn) {
  __shared__ float red[256];
  const int b = blockIdx.x, tid = threadIdx.x;
  const float* s = scores + (size_t)b * Tdim;
  float mx = -1e30f;
  for (int t = tid; t < Tdim; t += 256) mx = fmaxf(mx, s[t]);
  red[tid] = mx; __syncthreads();
  for (int o = 128; o > 0; o >>= 1) { if (tid < o) red[tid] = fmaxf(red[tid], red[tid + o]); __syncthreads(); }
  mx = red[0]; __syncthreads();
  float sum = 0.f;
  for (int t = tid; t < Tdim; t += 256) sum += expf(s[t] - mx);
  red[tid] = sum; __syncthreads();
  for (int o = 128; o > 0; o >>= 1) { if (tid < o) red[tid] += red[tid + o]; __syncthreads(); }
  sum = red[0];
  const float inv = 1.0f / sum;
  for (int t = tid; t < Tdim; t += 256) attn[(size_t)b * Tdim + t] = expf(s[t] - mx) * inv;
}

// context partials: ctx_part[b][tc][h] = sum_{t in chunk} attn[b][t]*enc[t*B+b][h]
__global__ __launch_bounds__(256) void k_context_part(const float* __restrict__ attn,
                                                      const float* __restrict__ enc,
                                                      float* __restrict__ ctx_part) {
  const int hidx = blockIdx.x * 256 + threadIdx.x;
  const int tc = blockIdx.y, b = blockIdx.z;
  float acc = 0.f;
  for (int t = tc * 64; t < tc * 64 + 64; ++t)
    acc += attn[(size_t)b * Tdim + t] * enc[((size_t)t * Bdim + b) * Hdim + hidx];
  ctx_part[((size_t)b * 64 + tc) * Hdim + hidx] = acc;
}

__global__ __launch_bounds__(256) void k_ctx_reduce(const float* __restrict__ ctx_part,
                                                    float* __restrict__ context) {
  const int idx = blockIdx.x * 256 + threadIdx.x;  // < B*H
  const int b = idx / Hdim, hh = idx % Hdim;
  float s = 0.f;
  for (int tc = 0; tc < 64; ++tc) s += ctx_part[((size_t)b * 64 + tc) * Hdim + hh];
  context[idx] = s;
}

// GRU pre-activations: gi = x@w_ih^T + b_ih, gh = h@w_hh^T + b_hh; wave per (b,j)
__global__ __launch_bounds__(256)
void k_gates(const int* __restrict__ token, const float* __restrict__ emb,
             const float* __restrict__ context, const float* __restrict__ hvec,
             const float* __restrict__ w_ih, const float* __restrict__ w_hh,
             const float* __restrict__ b_ih, const float* __restrict__ b_hh,
             float* __restrict__ gi, float* __restrict__ gh) {
  const int lane = threadIdx.x & 31;
  const int wid  = blockIdx.x * 8 + (threadIdx.x >> 5);
  if (wid >= Bdim * H3) return;
  const int b = wid / H3, j = wid % H3;
  const float* wi = w_ih + (size_t)j * Hdim;
  const float* wh = w_hh + (size_t)j * Hdim;
  const float* e  = emb + (size_t)token[b] * Hdim;
  const float* cx = context + b * Hdim;
  const float* hb = hvec + b * Hdim;
  float a0 = 0.f, a1 = 0.f;
  for (int k = lane; k < Hdim; k += 32) {
    a0 += (e[k] + cx[k]) * wi[k];
    a1 += hb[k] * wh[k];
  }
  a0 += __shfl_xor(a0, 1); a0 += __shfl_xor(a0, 2); a0 += __shfl_xor(a0, 4);
  a0 += __shfl_xor(a0, 8); a0 += __shfl_xor(a0, 16);
  a1 += __shfl_xor(a1, 1); a1 += __shfl_xor(a1, 2); a1 += __shfl_xor(a1, 4);
  a1 += __shfl_xor(a1, 8); a1 += __shfl_xor(a1, 16);
  if (lane == 0) {
    gi[(size_t)b * H3 + j] = a0 + b_ih[j];
    gh[(size_t)b * H3 + j] = a1 + b_hh[j];
  }
}

// GRU combine -> h_new; write to output tail and to zero-padded 16xH buffer
__global__ __launch_bounds__(256) void k_gru(const float* __restrict__ gi,
                                             const float* __restrict__ gh,
                                             const float* __restrict__ hvec,
                                             float* __restrict__ hn16,
                                             float* __restrict__ out) {
  const int idx = blockIdx.x * 256 + threadIdx.x;  // < 16*H
  const int b = idx >> 10, hh = idx & 1023;
  if (b < Bdim) {
    const float ir  = gi[(size_t)b * H3 + hh];
    const float iz  = gi[(size_t)b * H3 + Hdim + hh];
    const float in_ = gi[(size_t)b * H3 + 2 * Hdim + hh];
    const float hr  = gh[(size_t)b * H3 + hh];
    const float hz  = gh[(size_t)b * H3 + Hdim + hh];
    const float hn  = gh[(size_t)b * H3 + 2 * Hdim + hh];
    const float r = 1.0f / (1.0f + expf(-(ir + hr)));
    const float z = 1.0f / (1.0f + expf(-(iz + hz)));
    const float n = tanhf(in_ + r * hn);
    const float hprev = hvec[b * Hdim + hh];
    const float hnew = (1.0f - z) * n + z * hprev;
    hn16[idx] = hnew;
    out[(size_t)Bdim * Vdim + (size_t)b * Hdim + hh] = hnew;
  } else {
    hn16[idx] = 0.f;
  }
}

// ---------------------------------------------------------------------------
// classifier GEMM: logits = hn16(16xH) @ cls_w^T + cls_b; wave per 16-col tile.
// hn16 (64KB) staged once into LDS with async b128 copies; cls_w streamed.
// ---------------------------------------------------------------------------
__global__ __launch_bounds__(256)
void k_classifier(const float* __restrict__ hn16, const float* __restrict__ cls_w,
                  const float* __restrict__ cls_b, float* __restrict__ logits) {
  __shared__ __align__(16) float Ash[16 * Hdim];   // 64 KB

  const int tid  = threadIdx.x;
  const int lane = tid & 31;
  const int wave = tid >> 5;

  // stage hn16 -> LDS: 4096 x 16B chunks, 256 threads x 16 iterations
#pragma unroll
  for (int it = 0; it < 16; ++it) {
    const int chunk = it * 256 + tid;          // *4 floats
    const float* lp = &Ash[chunk * 4];
    const float* gp = hn16 + chunk * 4;
    ASYNC_LDS_B128(lp, gp);
  }
  WAIT_ASYNCCNT_0();
  __syncthreads();   // barrier before any early exit (no divergent barrier)

  const int NT = (Vdim + 15) / 16;  // 3142
  const int nt = blockIdx.x * 8 + wave;
  if (nt >= NT) return;
  const int half = lane >> 4, l16 = lane & 15;
  const int v0 = nt * 16;
  const int v  = v0 + l16;

  const float* Arow = &Ash[l16 * Hdim];
  const float* Wrow = cls_w + (size_t)((v < Vdim) ? v : 0) * Hdim;

  f32x8 c = f32x8{0};
  for (int k0 = 0; k0 < Hdim; k0 += 32) {
    if (k0 + 256 < Hdim) __builtin_prefetch(Wrow + k0 + 256, 0, 1);
    const int koff = k0 + half * 16;
    bf16x16 bf = cvt16_consecutive(Wrow + koff);     // streaming global
    bf16x16 af = make_a_frag(Arow + k0, half);       // from LDS
    c = __builtin_amdgcn_wmma_f32_16x16x32_bf16(
        false, af, false, bf, (short)0, c, false, false);
  }

#pragma unroll
  for (int r = 0; r < 8; ++r) {
    const int row = r + half * 8;
    if (row < Bdim && v < Vdim)
      logits[(size_t)row * Vdim + v] = c[r] + cls_b[v];
  }
}

// softmax over V, one block per batch element; writes decoder_output
__global__ __launch_bounds__(256) void k_softmax_V(const float* __restrict__ logits,
                                                   float* __restrict__ out) {
  __shared__ float red[256];
  const int b = blockIdx.x, tid = threadIdx.x;
  const float* s = logits + (size_t)b * Vdim;
  float mx = -1e30f;
  for (int v = tid; v < Vdim; v += 256) mx = fmaxf(mx, s[v]);
  red[tid] = mx; __syncthreads();
  for (int o = 128; o > 0; o >>= 1) { if (tid < o) red[tid] = fmaxf(red[tid], red[tid + o]); __syncthreads(); }
  mx = red[0]; __syncthreads();
  float sum = 0.f;
  for (int v = tid; v < Vdim; v += 256) sum += expf(s[v] - mx);
  red[tid] = sum; __syncthreads();
  for (int o = 128; o > 0; o >>= 1) { if (tid < o) red[tid] += red[tid + o]; __syncthreads(); }
  sum = red[0];
  const float inv = 1.0f / sum;
  for (int v = tid; v < Vdim; v += 256) out[(size_t)b * Vdim + v] = expf(s[v] - mx) * inv;
}

// ---------------------------------------------------------------------------

extern "C" void kernel_launch(void* const* d_in, const int* in_sizes, int n_in,
                              void* d_out, int out_size, void* d_ws, size_t ws_size,
                              hipStream_t stream) {
  const int*   token   = (const int*)d_in[0];
  const float* dec_h   = (const float*)d_in[1];
  const float* enc_in  = (const float*)d_in[2];
  const float* emb     = (const float*)d_in[3];
  const float* fc_h_w  = (const float*)d_in[4];
  const float* fc_h_b  = (const float*)d_in[5];
  const float* fc_e_w  = (const float*)d_in[6];
  const float* fc_e_b  = (const float*)d_in[7];
  const float* attn_w  = (const float*)d_in[8];
  const float* w_ih    = (const float*)d_in[9];
  const float* w_hh    = (const float*)d_in[10];
  const float* b_ih    = (const float*)d_in[11];
  const float* b_hh    = (const float*)d_in[12];
  const float* cls_w   = (const float*)d_in[13];
  const float* cls_b   = (const float*)d_in[14];
  float* out = (float*)d_out;

  size_t off = 0;
  auto wsalloc = [&](size_t nbytes) -> void* {
    void* p = (char*)d_ws + off;
    off = (off + nbytes + 255) & ~(size_t)255;
    return p;
  };
  float*          enc         = (float*)wsalloc(sizeof(float) * (size_t)Mdim * Hdim);
  unsigned short* Wb          = (unsigned short*)wsalloc(sizeof(unsigned short) * (size_t)Hdim * Hdim);
  float*          scores_part = (float*)wsalloc(sizeof(float) * (size_t)Mdim * 16);
  float*          scores      = (float*)wsalloc(sizeof(float) * (size_t)Bdim * Tdim);
  float*          attn        = (float*)wsalloc(sizeof(float) * (size_t)Bdim * Tdim);
  float*          hvec        = (float*)wsalloc(sizeof(float) * Bdim * Hdim);
  float*          ctx_part    = (float*)wsalloc(sizeof(float) * (size_t)Bdim * 64 * Hdim);
  float*          context     = (float*)wsalloc(sizeof(float) * Bdim * Hdim);
  float*          gi          = (float*)wsalloc(sizeof(float) * Bdim * H3);
  float*          gh          = (float*)wsalloc(sizeof(float) * Bdim * H3);
  float*          hn16        = (float*)wsalloc(sizeof(float) * 16 * Hdim);
  float*          logits      = (float*)wsalloc(sizeof(float) * (size_t)Bdim * Vdim);

  k_convert_w<<<(Hdim * Hdim) / 256, 256, 0, stream>>>(fc_e_w, Wb);
  k_fc_hidden<<<(Bdim * Hdim + 7) / 8, 256, 0, stream>>>(dec_h, fc_h_w, fc_h_b, hvec);
  k_gemm_enc<<<dim3(2, Mdim / 16), 256, 0, stream>>>(enc_in, Wb, fc_e_b, hvec, attn_w,
                                                     enc, scores_part);
  k_scores_reduce<<<Mdim / 256, 256, 0, stream>>>(scores_part, scores);
  k_softmax_T<<<Bdim, 256, 0, stream>>>(scores, attn);
  k_context_part<<<dim3(Hdim / 256, 64, Bdim), 256, 0, stream>>>(attn, enc, ctx_part);
  k_ctx_reduce<<<(Bdim * Hdim) / 256, 256, 0, stream>>>(ctx_part, context);
  k_gates<<<(Bdim * H3 + 7) / 8, 256, 0, stream>>>(token, emb, context, hvec,
                                                   w_ih, w_hh, b_ih, b_hh, gi, gh);
  k_gru<<<(16 * Hdim) / 256, 256, 0, stream>>>(gi, gh, hvec, hn16, out);
  {
    const int NT = (Vdim + 15) / 16;
    k_classifier<<<(NT + 7) / 8, 256, 0, stream>>>(hn16, cls_w, cls_b, logits);
  }
  k_softmax_V<<<Bdim, 256, 0, stream>>>(logits, out);
}
